// Model_21062519620317
// MI455X (gfx1250) — compile-verified, matrix operand order checked by reading, not run on hardware
//
#include <hip/hip_runtime.h>
#include <stdint.h>

#define NN 16384
#define TT 5
#define MM 512
#define DD 128

typedef __attribute__((ext_vector_type(2)))  float  v2f;
typedef __attribute__((ext_vector_type(8)))  float  v8f;
typedef __attribute__((ext_vector_type(16))) __bf16 v16bf;

union BF16Frag { v16bf v; uint32_t u[8]; };

#if defined(__has_builtin)
# if __has_builtin(__builtin_amdgcn_tensor_load_to_lds)
#  define HAVE_TDM 1
# else
#  define HAVE_TDM 0
# endif
# if __has_builtin(__builtin_amdgcn_wmma_f32_16x16x4_f32)
#  define HAVE_WMMA_F32X4 1
# else
#  define HAVE_WMMA_F32X4 0
# endif
#else
# define HAVE_TDM 0
# define HAVE_WMMA_F32X4 0
#endif

// ---------------- helpers ----------------

__device__ __forceinline__ uint16_t f2bf(float f) {
  uint32_t u = __float_as_uint(f);
  uint32_t r = u + 0x7fffu + ((u >> 16) & 1u);   // round-to-nearest-even
  return (uint16_t)(r >> 16);
}

// K-pair index inside a 16x32 bf16 A (or 32x16 B, symmetric) WMMA fragment.
// lanes 0-15 hold K-halves {0..7, 16..23}; lanes 16-31 hold {8..15, 24..31}.
__device__ __forceinline__ int pair_idx(int lane, int j, int kbase_pairs) {
  int half4 = ((lane >> 4) & 1) * 4;
  int p = (j < 4) ? (half4 + j) : (8 + half4 + (j - 4));
  return kbase_pairs + p;
}

#if HAVE_TDM
typedef __attribute__((ext_vector_type(4))) unsigned int tdm_u4;
typedef __attribute__((ext_vector_type(8))) int tdm_i8;
typedef __attribute__((ext_vector_type(4))) int tdm_i4;

// Build a D# for a simple 2D tile (contiguous into LDS) and issue TDM load.
// 6-arg toolchain variant: (u32x4 g0, i32x8 g1, i32x4, i32x4, i32x8, i32 cpol)
__device__ __forceinline__ void tdm_load_lds_2d(uint32_t lds_byte, const void* gptr,
                                                uint32_t cols, uint32_t rows,
                                                uint32_t row_stride_elems, int dsz_code) {
  uint64_t ga = (uint64_t)(uintptr_t)gptr;
  union { uint64_t q[2]; tdm_u4 v; } g0;
  g0.q[0] = 1ull /*count=1*/ | ((uint64_t)lds_byte << 32);
  g0.q[1] = (ga & ((1ull << 57) - 1)) | (2ull << 62); // type=2 ("image")
  union { uint64_t q[4]; tdm_i8 v; } g1;
  uint64_t td0 = row_stride_elems;   // tensor_dim0 (OOB extent)
  uint64_t td1 = rows;               // tensor_dim1
  uint64_t t0  = cols;               // tile_dim0
  uint64_t t1  = rows;               // tile_dim1
  uint64_t s0  = row_stride_elems;   // tensor_dim0_stride
  g1.q[0] = ((uint64_t)(dsz_code & 3) << 16) | ((td0 & 0xffffull) << 48);
  g1.q[1] = ((td0 >> 16) & 0xffffull) | ((td1 & 0xffffffffull) << 16) | ((t0 & 0xffffull) << 48);
  g1.q[2] = (t1 & 0xffffull) | ((s0 & 0xffffffffull) << 32);
  g1.q[3] = (s0 >> 32) & 0xffffull;  // tensor_dim1_stride = 0 (2D tile)
  tdm_i4 z4 = {0, 0, 0, 0};
  tdm_i8 z8 = {0, 0, 0, 0, 0, 0, 0, 0};
  __builtin_amdgcn_tensor_load_to_lds(g0.v, g1.v, z4, z4, z8, 0);
}
#endif

// Stage a contiguous bf16 block into LDS (TDM when available).
__device__ __forceinline__ void stage_lds_bf16(uint16_t* sdst, const uint16_t* gsrc, int nelem) {
#if HAVE_TDM
  if (threadIdx.x < 32) {
    tdm_load_lds_2d((uint32_t)(uintptr_t)sdst, gsrc, (uint32_t)nelem, 1u, (uint32_t)nelem, 1);
    __builtin_amdgcn_s_wait_tensorcnt(0);
  }
  __syncthreads();
#else
  for (int i = threadIdx.x; i < nelem; i += blockDim.x) sdst[i] = gsrc[i];
  __syncthreads();
#endif
}

// Distributed top-32 (one candidate per lane of a wave32), smaller == better.
__device__ __forceinline__ void topk_insert(float c, int idx, float& bv, int& bi,
                                            float& vmax, int lane) {
  if (c < vmax) {
    unsigned long long m = __ballot(bv == vmax);
    int first = __ffsll(m) - 1;
    if (lane == first) { bv = c; bi = idx; }
    float r = bv;
    #pragma unroll
    for (int off = 16; off > 0; off >>= 1) r = fmaxf(r, __shfl_xor(r, off, 32));
    vmax = r;
  }
}

// Emit the 32 kept candidates in ascending order into dst[0..31] (LDS).
__device__ __forceinline__ void topk_emit(float bv, int bi, int lane, int* dst) {
  float v = bv; int id = bi;
  for (int k = 0; k < 32; ++k) {
    float r = v;
    #pragma unroll
    for (int off = 16; off > 0; off >>= 1) r = fminf(r, __shfl_xor(r, off, 32));
    unsigned long long m = __ballot(v == r);
    int first = __ffsll(m) - 1;
    int widx = __shfl(id, first, 32);
    if (lane == 0) dst[k] = widx;
    if (lane == first) v = INFINITY;
  }
}

// ---------------- workspace layout (bytes) ----------------
constexpr size_t WS_AIR   = 0;                       // 512 float4        (8 KB)
constexpr size_t WS_FLAGS = WS_AIR   + 8192;         // N int             (64 KB)
constexpr size_t WS_POSA  = WS_FLAGS + 65536;        // N float4          (256 KB)
constexpr size_t WS_POSB  = WS_POSA  + 262144;       // N float4          (256 KB)
constexpr size_t WS_XBF   = WS_POSB  + 262144;       // N x 32 bf16       (1 MB)
constexpr size_t WS_WENCT = WS_XBF   + 1048576;      // 128x32 bf16
constexpr size_t WS_WM1T  = WS_WENCT + 8192;         // 128x128 bf16
constexpr size_t WS_WM2T  = WS_WM1T  + 32768;        // 128x128 bf16
constexpr size_t WS_WDECT = WS_WM2T  + 32768;        // 16x128 bf16
constexpr size_t WS_H     = WS_WDECT + 4096;         // N x 128 f32       (8 MB)
constexpr size_t WS_AGG   = WS_H     + 8388608;      // N x 128 bf16      (4 MB)
constexpr size_t WS_NBR   = WS_AGG   + 4194304;      // N x 32 int        (2 MB)

// ---------------- kernels ----------------

// Gather airfoil positions, set flags, convert weights to transposed bf16.
__global__ void __launch_bounds__(512) k_prep(
    const float* __restrict__ pos, const int* __restrict__ idx,
    const float* __restrict__ Wenc, const float* __restrict__ Wm1,
    const float* __restrict__ Wm2, const float* __restrict__ Wdec,
    float* air, int* flags, uint16_t* WencT, uint16_t* Wm1T,
    uint16_t* Wm2T, uint16_t* WdecT) {
  int tid = threadIdx.x;
  for (int i = tid; i < NN; i += 512) flags[i] = 0;
  __syncthreads();
  if (tid < MM) {
    int n = idx[tid];
    flags[n] = 1;
    air[tid * 4 + 0] = pos[n * 3 + 0];
    air[tid * 4 + 1] = pos[n * 3 + 1];
    air[tid * 4 + 2] = pos[n * 3 + 2];
    air[tid * 4 + 3] = 0.f;
  }
  for (int e = tid; e < DD * 32; e += 512) {
    int c = e >> 5, k = e & 31;
    WencT[e] = (k < TT * 3 + 2) ? f2bf(Wenc[k * DD + c]) : (uint16_t)0;
  }
  for (int e = tid; e < DD * DD; e += 512) {
    int c = e >> 7, k = e & 127;
    Wm1T[e] = f2bf(Wm1[k * DD + c]);
    Wm2T[e] = f2bf(Wm2[k * DD + c]);
  }
  for (int e = tid; e < 16 * DD; e += 512) {
    int c = e >> 7, k = e & 127;
    WdecT[e] = (c < TT * 3) ? f2bf(Wdec[k * (TT * 3) + c]) : (uint16_t)0;
  }
}

// Wall distance + normalized positions (KNN operands) + bf16 feature rows.
__global__ void __launch_bounds__(256) k_features(
    const float* __restrict__ pos, const float* __restrict__ vel,
    const float* __restrict__ air, const int* __restrict__ flags,
    float* posA, float* posB, uint16_t* xbf) {
  __shared__ float4 sAir[MM];
  int tid = threadIdx.x;
  const float4* a4 = (const float4*)air;
  for (int i = tid; i < MM; i += 256) sAir[i] = a4[i];
  __syncthreads();
  int n = blockIdx.x * 256 + tid;
  float px = pos[n * 3 + 0], py = pos[n * 3 + 1], pz = pos[n * 3 + 2];
  float dmin = INFINITY;
  for (int j = 0; j < MM; ++j) {
    float4 q = sAir[j];
    float dx = px - q.x, dy = py - q.y, dz = pz - q.z;
    dmin = fminf(dmin, dx * dx + dy * dy + dz * dz);
  }
  float wall = sqrtf(dmin);
  float nx = (px - 0.8507418036460876f) / (0.40274253487586975f + 1e-8f);
  float ny = (py - (-6.422636200653642e-09f)) / (0.07883177697658539f + 1e-8f);
  float nz = (pz - 0.37120404839515686f) / (0.2320450097322464f + 1e-8f);
  float sq = nx * nx + ny * ny + nz * nz;
  posA[n * 4 + 0] = -2.f * nx; posA[n * 4 + 1] = -2.f * ny;
  posA[n * 4 + 2] = -2.f * nz; posA[n * 4 + 3] = sq;
  posB[n * 4 + 0] = nx; posB[n * 4 + 1] = ny;
  posB[n * 4 + 2] = nz; posB[n * 4 + 3] = 1.f;
  uint16_t* xr = xbf + (size_t)n * 32;
  #pragma unroll
  for (int t = 0; t < TT; ++t) {
    #pragma unroll
    for (int c = 0; c < 3; ++c) {
      float vm = (c == 0) ? 37.750118255615234f : ((c == 1) ? 0.5372318625450134f : 2.009599447250366f);
      float vs = (c == 0) ? 19.8649845123291f   : ((c == 1) ? 7.343273639678955f  : 9.551141738891602f);
      float v = (vel[((size_t)t * NN + n) * 3 + c] - vm) / (vs + 1e-8f);
      xr[t * 3 + c] = f2bf(v);
    }
  }
  xr[15] = f2bf(wall);
  xr[16] = f2bf((float)flags[n]);
  #pragma unroll
  for (int k = 17; k < 32; ++k) xr[k] = 0;
}

// Brute-force KNN: distance tiles via V_WMMA_F32_16X16X4_F32, top-32 per row.
__global__ void __launch_bounds__(256) k_knn(
    const float* __restrict__ posA, const float* __restrict__ posB, int* nbr) {
  __shared__ float4 sA[16];
  __shared__ float4 sB[256];
  __shared__ float sDist[16 * 256];
  __shared__ int sOut[16 * 32];
  int tid = threadIdx.x, lane = tid & 31, wave = tid >> 5;
  int rowbase = blockIdx.x * 16;
  if (tid < 16) sA[tid] = ((const float4*)posA)[rowbase + tid];
  __syncthreads();

  int K0 = (lane >> 4) << 1;  // lanes 0-15: K=0,1 ; lanes 16-31: K=2,3
  const float* ap = (const float*)&sA[lane & 15];
  v2f afrag; afrag.x = ap[K0]; afrag.y = ap[K0 + 1];

  float bv0 = INFINITY, bv1 = INFINITY, vmax0 = INFINITY, vmax1 = INFINITY;
  int bi0 = -1, bi1 = -1;
  int lr0 = wave * 2, lr1 = wave * 2 + 1;

  for (int cb = 0; cb < NN / 256; ++cb) {
    int colbase = cb * 256;
    sB[tid] = ((const float4*)posB)[colbase + tid];
    __syncthreads();
    #pragma unroll
    for (int tt = 0; tt < 2; ++tt) {
      int lcol = wave * 32 + tt * 16 + (lane & 15);
      const float* bp = (const float*)&sB[lcol];
      float sqc = bp[0] * bp[0] + bp[1] * bp[1] + bp[2] * bp[2];
#if HAVE_WMMA_F32X4
      v2f bfrag; bfrag.x = bp[K0]; bfrag.y = bp[K0 + 1];
      v8f c = {};
      c = __builtin_amdgcn_wmma_f32_16x16x4_f32(false, afrag, false, bfrag,
                                                (short)0, c, false, false);
      #pragma unroll
      for (int v = 0; v < 8; ++v) {
        int rm = v + ((lane >> 4) << 3);
        float d = c[v] + sqc;
        if (rowbase + rm == colbase + lcol) d = INFINITY;
        sDist[rm * 256 + lcol] = d;
      }
#else
      #pragma unroll
      for (int v = 0; v < 8; ++v) {
        int rm = v + ((lane >> 4) << 3);
        const float* ar = (const float*)&sA[rm];
        float d = ar[0] * bp[0] + ar[1] * bp[1] + ar[2] * bp[2] + ar[3] * bp[3] + sqc;
        if (rowbase + rm == colbase + lcol) d = INFINITY;
        sDist[rm * 256 + lcol] = d;
      }
#endif
    }
    __syncthreads();
    for (int j = 0; j < 256; ++j) {
      float c0 = sDist[lr0 * 256 + j];
      float c1 = sDist[lr1 * 256 + j];
      topk_insert(c0, colbase + j, bv0, bi0, vmax0, lane);
      topk_insert(c1, colbase + j, bv1, bi1, vmax1, lane);
    }
    __syncthreads();
  }
  topk_emit(bv0, bi0, lane, &sOut[lr0 * 32]);
  topk_emit(bv1, bi1, lane, &sOut[lr1 * 32]);
  __syncthreads();
  for (int i = tid; i < 16 * 32; i += 256)
    nbr[(size_t)(rowbase + (i >> 5)) * 32 + (i & 31)] = sOut[i];
}

// Encoder: relu(X[N,32bf] @ WencT + b) -> h[N,128] f32. One WMMA per wave.
__global__ void __launch_bounds__(256) k_encoder(
    const uint16_t* __restrict__ xbf, const uint16_t* __restrict__ WencT,
    const float* __restrict__ benc, float* h) {
  __shared__ uint16_t sW[DD * 32];
  stage_lds_bf16(sW, WencT, DD * 32);
  int lane = threadIdx.x & 31, wave = threadIdx.x >> 5;
  int rowbase = blockIdx.x * 16, colbase = wave * 16;
  int row = rowbase + (lane & 15);
  int col = colbase + (lane & 15);
  const uint32_t* A32 = (const uint32_t*)xbf;
  const uint32_t* W32 = (const uint32_t*)sW;
  BF16Frag a, b;
  #pragma unroll
  for (int j = 0; j < 8; ++j) {
    a.u[j] = A32[row * 16 + pair_idx(lane, j, 0)];
    b.u[j] = W32[col * 16 + pair_idx(lane, j, 0)];
  }
  v8f c = {};
  c = __builtin_amdgcn_wmma_f32_16x16x32_bf16(false, a.v, false, b.v, (short)0, c, false, false);
  float bias = benc[col];
  #pragma unroll
  for (int v = 0; v < 8; ++v) {
    int rm = rowbase + v + ((lane >> 4) << 3);
    float val = c[v] + bias;
    h[(size_t)rm * DD + col] = val > 0.f ? val : 0.f;
  }
}

// Mean aggregation over k neighbors -> bf16 rows for the message GEMM.
__global__ void __launch_bounds__(128) k_agg(
    const float* __restrict__ h, const int* __restrict__ nbr,
    uint16_t* aggbf, int k) {
  __shared__ int nb[32];
  int n = blockIdx.x, tid = threadIdx.x;
  if (tid < k) nb[tid] = nbr[(size_t)n * 32 + tid];
  __syncthreads();
  float s = 0.f;
  for (int j = 0; j < k; ++j) {
    if (j + 1 < k) __builtin_prefetch(&h[(size_t)nb[j + 1] * DD + tid], 0, 0);
    s += h[(size_t)nb[j] * DD + tid];
  }
  aggbf[(size_t)n * DD + tid] = f2bf(s / (float)k);
}

// h += relu(agg[N,128]bf @ WT[128,128]bf). W staged in LDS via TDM.
__global__ void __launch_bounds__(256) k_gemm_add(
    const uint16_t* __restrict__ aggbf, const uint16_t* __restrict__ WT, float* h) {
  __shared__ uint16_t sW[DD * DD];
  stage_lds_bf16(sW, WT, DD * DD);
  int lane = threadIdx.x & 31, wave = threadIdx.x >> 5;
  int rowbase = blockIdx.x * 16, colbase = wave * 16;
  int row = rowbase + (lane & 15);
  int col = colbase + (lane & 15);
  const uint32_t* A32 = (const uint32_t*)aggbf;
  const uint32_t* W32 = (const uint32_t*)sW;
  v8f c = {};
  #pragma unroll
  for (int kk = 0; kk < 4; ++kk) {
    BF16Frag a, b;
    #pragma unroll
    for (int j = 0; j < 8; ++j) {
      int p = pair_idx(lane, j, kk * 16);
      a.u[j] = A32[row * 64 + p];
      b.u[j] = W32[col * 64 + p];
    }
    c = __builtin_amdgcn_wmma_f32_16x16x32_bf16(false, a.v, false, b.v, (short)0, c, false, false);
  }
  #pragma unroll
  for (int v = 0; v < 8; ++v) {
    int rm = rowbase + v + ((lane >> 4) << 3);
    float val = c[v];
    val = val > 0.f ? val : 0.f;
    h[(size_t)rm * DD + col] += val;
  }
}

// Final hidden state -> bf16 for the decoder GEMM.
__global__ void __launch_bounds__(256) k_h2bf(const float* __restrict__ h, uint16_t* hbf) {
  size_t i = (size_t)blockIdx.x * 256 + threadIdx.x;
  hbf[i] = f2bf(h[i]);
}

// Decoder: h @ Wdec -> denormalize, zero airfoil rows, scatter to [T,N,3].
__global__ void __launch_bounds__(32) k_decoder(
    const uint16_t* __restrict__ hbf, const uint16_t* __restrict__ WdecT,
    const int* __restrict__ flags, float* out) {
  __shared__ uint16_t sW[16 * DD];
  stage_lds_bf16(sW, WdecT, 16 * DD);
  int lane = threadIdx.x & 31;
  int rowbase = blockIdx.x * 16;
  int row = rowbase + (lane & 15);
  int col = lane & 15;
  const uint32_t* A32 = (const uint32_t*)hbf;
  const uint32_t* W32 = (const uint32_t*)sW;
  v8f c = {};
  #pragma unroll
  for (int kk = 0; kk < 4; ++kk) {
    BF16Frag a, b;
    #pragma unroll
    for (int j = 0; j < 8; ++j) {
      int p = pair_idx(lane, j, kk * 16);
      a.u[j] = A32[row * 64 + p];
      b.u[j] = W32[col * 64 + p];
    }
    c = __builtin_amdgcn_wmma_f32_16x16x32_bf16(false, a.v, false, b.v, (short)0, c, false, false);
  }
  if (col < TT * 3) {
    int t = col / 3, ch = col % 3;
    float vm = (ch == 0) ? 37.750118255615234f : ((ch == 1) ? 0.5372318625450134f : 2.009599447250366f);
    float vs = (ch == 0) ? 19.8649845123291f   : ((ch == 1) ? 7.343273639678955f  : 9.551141738891602f);
    #pragma unroll
    for (int v = 0; v < 8; ++v) {
      int rm = rowbase + v + ((lane >> 4) << 3);
      float val = c[v] * vs + vm;
      if (flags[rm]) val = 0.f;
      out[((size_t)t * NN + rm) * 3 + ch] = val;
    }
  }
}

// ---------------- launch ----------------

extern "C" void kernel_launch(void* const* d_in, const int* in_sizes, int n_in,
                              void* d_out, int out_size, void* d_ws, size_t ws_size,
                              hipStream_t stream) {
  (void)in_sizes; (void)n_in; (void)out_size; (void)ws_size;
  const float* pos  = (const float*)d_in[1];
  const int*   idx  = (const int*)d_in[2];
  const float* vel  = (const float*)d_in[3];
  const float* Wenc = (const float*)d_in[4];
  const float* benc = (const float*)d_in[5];
  const float* Wm1  = (const float*)d_in[6];
  const float* Wm2  = (const float*)d_in[7];
  const float* Wdec = (const float*)d_in[8];
  float* out = (float*)d_out;
  char* ws = (char*)d_ws;
  float*    air   = (float*)(ws + WS_AIR);
  int*      flags = (int*)(ws + WS_FLAGS);
  float*    posA  = (float*)(ws + WS_POSA);
  float*    posB  = (float*)(ws + WS_POSB);
  uint16_t* xbf   = (uint16_t*)(ws + WS_XBF);
  uint16_t* WencT = (uint16_t*)(ws + WS_WENCT);
  uint16_t* Wm1T  = (uint16_t*)(ws + WS_WM1T);
  uint16_t* Wm2T  = (uint16_t*)(ws + WS_WM2T);
  uint16_t* WdecT = (uint16_t*)(ws + WS_WDECT);
  float*    h     = (float*)(ws + WS_H);
  uint16_t* agg   = (uint16_t*)(ws + WS_AGG);
  int*      nbr   = (int*)(ws + WS_NBR);

  k_prep<<<1, 512, 0, stream>>>(pos, idx, Wenc, Wm1, Wm2, Wdec,
                                air, flags, WencT, Wm1T, Wm2T, WdecT);
  k_features<<<NN / 256, 256, 0, stream>>>(pos, vel, air, flags, posA, posB, xbf);
  k_knn<<<NN / 16, 256, 0, stream>>>(posA, posB, nbr);
  k_encoder<<<NN / 16, 256, 0, stream>>>(xbf, WencT, benc, h);
  k_agg<<<NN, 128, 0, stream>>>(h, nbr, agg, 16);
  k_gemm_add<<<NN / 16, 256, 0, stream>>>(agg, Wm1T, h);
  k_agg<<<NN, 128, 0, stream>>>(h, nbr, agg, 32);
  k_gemm_add<<<NN / 16, 256, 0, stream>>>(agg, Wm2T, h);
  k_h2bf<<<(NN * DD) / 256, 256, 0, stream>>>(h, agg);
  k_decoder<<<NN / 16, 32, 0, stream>>>(agg, WdecT, flags, out);
}
